// InteractionBlock_2259152798300
// MI455X (gfx1250) — matrix operand order repaired and checked by reading, not compiled, not run on hardware
//
#include <hip/hip_runtime.h>
#include <hip/hip_bf16.h>

typedef __attribute__((ext_vector_type(16))) _Float16 v16h;
typedef __attribute__((ext_vector_type(8)))  _Float16 v8h;
typedef __attribute__((ext_vector_type(8)))  float    v8f;
typedef __attribute__((ext_vector_type(4)))  float    v4f;

#define N_NODES 10000
#define N_EDGES 320000
#define CH 64
#define R_DIM 8
#define PLANE (N_NODES * CH)   // one (comp) plane of agg

__device__ __forceinline__ float silu_f(float x) {
    return x / (1.0f + __expf(-x));
}

__device__ __forceinline__ v16h cat8(v8h lo, v8h hi) {
    return __builtin_shufflevector(lo, hi, 0, 1, 2, 3, 4, 5, 6, 7,
                                   8, 9, 10, 11, 12, 13, 14, 15);
}

// A fragment (16x32 f16) from a row-major f16 tile in LDS.
// Lane layout (ISA 7.12.2): row m = lane&15; elements 0..7 = K hl*8..hl*8+7,
// elements 8..15 = K 16+hl*8..16+hl*8+7  ->  two contiguous 16B runs.
__device__ __forceinline__ v16h load_a_lds(const _Float16* base, int stride, int lane) {
    int m = lane & 15, hl = lane >> 4;
    const _Float16* p = base + m * stride + (hl << 3);
    v8h lo = *(const v8h*)p;
    v8h hi = *(const v8h*)(p + 16);
    return cat8(lo, hi);
}

// Same A fragment sourced from row-major f32 in global memory (convert to f16).
__device__ __forceinline__ v16h load_a_f32(const float* base, int stride, int lane) {
    int m = lane & 15, hl = lane >> 4;
    const float* p = base + m * stride + (hl << 3);
    v8f lo = *(const v8f*)p;
    v8f hi = *(const v8f*)(p + 16);
    v16h a;
#pragma unroll
    for (int i = 0; i < 8; ++i) {
        a[i]     = (_Float16)lo[i];
        a[i + 8] = (_Float16)hi[i];
    }
    return a;
}

// B fragment from LDS pre-swizzled into fragment order: one 32B contiguous read.
__device__ __forceinline__ v16h load_b_sw(const _Float16* frag, int lane) {
    return *(const v16h*)(frag + (lane << 4));
}

__device__ __forceinline__ v8f wmma_f16(v16h a, v16h b, v8f c) {
    return __builtin_amdgcn_wmma_f32_16x16x32_f16(false, a, false, b, (short)0, c, false, false);
}

// Convert f32 weights [Krows x Ncols] (global) into fragment-ordered f16 LDS:
// dst[((kt*NT + nt)*32 + lane)*16 + i] = W[(kt*32 + (i>>3)*16 + (lane>>4)*8 + (i&7)) * Ncols + nt*16 + (lane&15)]
// (B element i of a lane maps to k_local = (i>>3)*16 + hl*8 + (i&7), matching the A layout)
__device__ __forceinline__ void fill_bsw(_Float16* dst, const float* W,
                                         int Krows, int Ncols, int KT, int NT,
                                         int tid, int nthreads) {
    int total = KT * NT * 512;
    for (int f = tid; f < total; f += nthreads) {
        int i = f & 15;
        int lane = (f >> 4) & 31;
        int t = f >> 9;
        int nt = t % NT, kt = t / NT;
        int k = kt * 32 + ((i >> 3) << 4) + ((lane >> 4) << 3) + (i & 7);
        int n = nt * 16 + (lane & 15);
        dst[f] = (k < Krows) ? (_Float16)W[k * Ncols + n] : (_Float16)0.0f;
    }
}

// ---------------------------------------------------------------------------
// Kernel 0: zero agg planes (ws is poisoned)
// ---------------------------------------------------------------------------
__global__ void zero_kernel(float* p, int count) {
    int stride = gridDim.x * blockDim.x;
    for (int i = blockIdx.x * blockDim.x + threadIdx.x; i < count; i += stride)
        p[i] = 0.0f;
}

// ---------------------------------------------------------------------------
// Kernel 1: h = (node_feats @ W_up) / sqrt(CH)
// ---------------------------------------------------------------------------
__global__ void __launch_bounds__(256) up_kernel(const float* __restrict__ nf,
                                                 const float* __restrict__ Wup,
                                                 float* __restrict__ hG) {
    __shared__ __align__(32) _Float16 sW[2 * 4 * 512];  // KT=2, NT=4
    int tid = threadIdx.x;
    fill_bsw(sW, Wup, 64, 64, 2, 4, tid, 256);
    __syncthreads();

    int lane = tid & 31, wave = tid >> 5;
    int tile = blockIdx.x * 8 + wave;
    if (tile >= (N_NODES / 16)) return;
    int n0 = tile * 16;
    int hl = lane >> 4, m = lane & 15;

    v16h a0 = load_a_f32(nf + (long)n0 * 64, 64, lane);
    v16h a1 = load_a_f32(nf + (long)n0 * 64 + 32, 64, lane);

    v8f acc[4];
#pragma unroll
    for (int nt = 0; nt < 4; ++nt) { v8f z = {}; acc[nt] = z; }
#pragma unroll
    for (int nt = 0; nt < 4; ++nt)
        acc[nt] = wmma_f16(a0, load_b_sw(sW + (nt << 9), lane), acc[nt]);
#pragma unroll
    for (int nt = 0; nt < 4; ++nt)
        acc[nt] = wmma_f16(a1, load_b_sw(sW + ((4 + nt) << 9), lane), acc[nt]);

#pragma unroll
    for (int nt = 0; nt < 4; ++nt)
#pragma unroll
        for (int r = 0; r < 8; ++r) {
            int row = r + (hl << 3);
            hG[(long)(n0 + row) * 64 + nt * 16 + m] = acc[nt][r] * 0.125f;
        }
}

// ---------------------------------------------------------------------------
// Kernel 2: fused radial MLP (WMMA f16) + tensor-product message + scatter
// block = 128 threads = 4 waves; 16 edges per wave (exact tiling)
// agg layout: 9 planes of [N_NODES][64]
// ---------------------------------------------------------------------------
__global__ void __launch_bounds__(128) edge_kernel(const float* __restrict__ ev,
                                                   const float* __restrict__ radial,
                                                   const int* __restrict__ senders,
                                                   const int* __restrict__ receivers,
                                                   const float* __restrict__ W1,
                                                   const float* __restrict__ W2,
                                                   const float* __restrict__ W3,
                                                   const float* __restrict__ W4,
                                                   const float* __restrict__ hG,
                                                   float* __restrict__ agg) {
    __shared__ __align__(32) _Float16 sW1[1 * 4 * 512];    // K 8 -> pad 32
    __shared__ __align__(32) _Float16 sW2[2 * 4 * 512];
    __shared__ __align__(32) _Float16 sW3[2 * 4 * 512];
    __shared__ __align__(32) _Float16 sW4[2 * 12 * 512];
    __shared__ __align__(32) _Float16 sAct[4][16 * 64];    // row-major activations
    __shared__ __align__(32) _Float16 sHT[4][64 * 16];     // h[senders], [ch][row]
    __shared__ __align__(32) float    sSHT[4][9 * 16];     // sh, [comp][row]
    __shared__ int sSnd[4][16];
    __shared__ int sRcv[4][16];

    int tid = threadIdx.x;
    int lane = tid & 31, wave = tid >> 5;
    int hl = lane >> 4, m = lane & 15;
    long e0 = ((long)blockIdx.x * 4 + wave) * 16;

    // --- weights -> fragment-ordered f16 LDS (block cooperative) ---
    fill_bsw(sW1, W1, R_DIM, 64, 1, 4, tid, 128);
    fill_bsw(sW2, W2, 64, 64, 2, 4, tid, 128);
    fill_bsw(sW3, W3, 64, 64, 2, 4, tid, 128);
    fill_bsw(sW4, W4, 64, 192, 2, 12, tid, 128);

    // --- per-wave: edge ids, spherical harmonics (lanes 0..15) ---
    if (lane < 16) {
        long e = e0 + lane;
        sSnd[wave][lane] = senders[e];
        sRcv[wave][lane] = receivers[e];
        float x = ev[e * 3 + 0], y = ev[e * 3 + 1], z = ev[e * 3 + 2];
        float rn = rsqrtf(x * x + y * y + z * z);
        x *= rn; y *= rn; z *= rn;
        const float c3 = 1.7320508075688772f;
        const float c15 = 3.872983346207417f;
        const float c5 = 2.23606797749979f;
        float* o = &sSHT[wave][0];
        o[0 * 16 + lane] = 1.0f;
        o[1 * 16 + lane] = c3 * x;
        o[2 * 16 + lane] = c3 * y;
        o[3 * 16 + lane] = c3 * z;
        o[4 * 16 + lane] = c15 * x * y;
        o[5 * 16 + lane] = c15 * y * z;
        o[6 * 16 + lane] = 0.5f * c5 * (3.0f * z * z - 1.0f);
        o[7 * 16 + lane] = c15 * x * z;
        o[8 * 16 + lane] = 0.5f * c15 * (x * x - y * y);
    }

    // --- per-wave: gather h[senders] transposed into sHT[ch][row] ---
    {
        int row = m;                       // lane&15 -> edge row
        int snd = sSnd[wave][row];
        const float* hrow = hG + (long)snd * 64 + (hl << 5);  // half the channels
#pragma unroll
        for (int g = 0; g < 8; ++g) {
            v4f hv = *(const v4f*)(hrow + g * 4);
#pragma unroll
            for (int j = 0; j < 4; ++j) {
                int ch = (hl << 5) + g * 4 + j;
                sHT[wave][ch * 16 + row] = (_Float16)hv[j];
            }
        }
    }
    __syncthreads();

    // --- layer 1: [16 x 8(pad 32)] @ [32 x 64], scale 1/sqrt(8), silu ---
    v16h aF;
    {
        v8f r = *(const v8f*)(radial + (e0 + m) * R_DIM);
        float msk = (hl == 0) ? 1.0f : 0.0f;
#pragma unroll
        for (int i = 0; i < 8; ++i) aF[i] = (_Float16)(r[i] * msk);
#pragma unroll
        for (int i = 8; i < 16; ++i) aF[i] = (_Float16)0.0f;
    }
    {
        v8f acc[4];
#pragma unroll
        for (int nt = 0; nt < 4; ++nt) {
            v8f z = {};
            acc[nt] = wmma_f16(aF, load_b_sw(sW1 + (nt << 9), lane), z);
        }
        const float s1 = 0.35355339059327373f;
#pragma unroll
        for (int nt = 0; nt < 4; ++nt)
#pragma unroll
            for (int r = 0; r < 8; ++r) {
                int row = r + (hl << 3);
                sAct[wave][row * 64 + nt * 16 + m] = (_Float16)silu_f(acc[nt][r] * s1);
            }
    }

    // --- layers 2 & 3: [16 x 64] @ [64 x 64], scale 1/8, silu ---
    const _Float16* Ws[2] = { sW2, sW3 };
#pragma unroll
    for (int L = 0; L < 2; ++L) {
        v16h a0 = load_a_lds(&sAct[wave][0], 64, lane);
        v16h a1 = load_a_lds(&sAct[wave][32], 64, lane);
        v8f acc[4];
#pragma unroll
        for (int nt = 0; nt < 4; ++nt) {
            v8f z = {};
            acc[nt] = wmma_f16(a0, load_b_sw(Ws[L] + (nt << 9), lane), z);
        }
#pragma unroll
        for (int nt = 0; nt < 4; ++nt)
            acc[nt] = wmma_f16(a1, load_b_sw(Ws[L] + ((4 + nt) << 9), lane), acc[nt]);
#pragma unroll
        for (int nt = 0; nt < 4; ++nt)
#pragma unroll
            for (int r = 0; r < 8; ++r) {
                int row = r + (hl << 3);
                sAct[wave][row * 64 + nt * 16 + m] = (_Float16)silu_f(acc[nt][r] * 0.125f);
            }
    }

    // --- layer 4 + message + scatter ---
    {
        v16h a0 = load_a_lds(&sAct[wave][0], 64, lane);
        v16h a1 = load_a_lds(&sAct[wave][32], 64, lane);

        int rcvr[8];
#pragma unroll
        for (int r = 0; r < 8; ++r) rcvr[r] = sRcv[wave][r + (hl << 3)];

        v8h hv[4];
#pragma unroll
        for (int ntl = 0; ntl < 4; ++ntl) {
            int ch = (ntl << 4) + m;
            hv[ntl] = *(const v8h*)(&sHT[wave][ch * 16 + (hl << 3)]);
        }

#pragma unroll
        for (int l = 0; l < 3; ++l) {
            const int d = 2 * l + 1;
            const int moff = l * l;
            v8f shv[5];
#pragma unroll
            for (int c = 0; c < 5; ++c)
                if (c < d)
                    shv[c] = *(const v8f*)(&sSHT[wave][(moff + c) * 16 + (hl << 3)]);

#pragma unroll
            for (int ntl = 0; ntl < 4; ++ntl) {
                int nt = l * 4 + ntl;
                v8f c = {};
                c = wmma_f16(a0, load_b_sw(sW4 + (nt << 9), lane), c);
                c = wmma_f16(a1, load_b_sw(sW4 + ((12 + nt) << 9), lane), c);
                int ch = (ntl << 4) + m;
#pragma unroll
                for (int r = 0; r < 8; ++r) {
                    float hw = (float)hv[ntl][r] * (c[r] * 0.125f);
                    long node = (long)rcvr[r] * 64 + ch;
#pragma unroll
                    for (int mm = 0; mm < d; ++mm)
                        atomicAdd(agg + (long)(moff + mm) * PLANE + node,
                                  hw * shv[mm][r]);
                }
            }
        }
    }
}

// ---------------------------------------------------------------------------
// Kernel 3: out[n, d, m] = sum_c agg[m][n][c] * W_down[l][c, d] / 256
// ---------------------------------------------------------------------------
__global__ void __launch_bounds__(256) down_kernel(const float* __restrict__ agg,
                                                   const float* __restrict__ Wd,
                                                   float* __restrict__ out) {
    __shared__ __align__(32) _Float16 sWd[3 * 2 * 4 * 512];
    int tid = threadIdx.x;
#pragma unroll
    for (int l = 0; l < 3; ++l)
        fill_bsw(sWd + l * 4096, Wd + l * 4096, 64, 64, 2, 4, tid, 256);
    __syncthreads();

    int lane = tid & 31, wave = tid >> 5;
    int tile = blockIdx.x * 8 + wave;
    if (tile >= (N_NODES / 16)) return;
    int n0 = tile * 16;
    int hl = lane >> 4, m = lane & 15;

    const int lbase[3] = { 0, 64, 256 };
    const float scale = 1.0f / 256.0f;   // (1/AVG_NEIGH)*(1/sqrt(CH))

#pragma unroll
    for (int l = 0; l < 3; ++l) {
        const int d = 2 * l + 1;
        const int moff = l * l;
#pragma unroll
        for (int comp = 0; comp < d; ++comp) {
            const float* plane = agg + (long)(moff + comp) * PLANE + (long)n0 * 64;
            v16h a0 = load_a_f32(plane, 64, lane);
            v16h a1 = load_a_f32(plane + 32, 64, lane);
#pragma unroll
            for (int nt = 0; nt < 4; ++nt) {
                v8f c = {};
                c = wmma_f16(a0, load_b_sw(sWd + l * 4096 + (nt << 9), lane), c);
                c = wmma_f16(a1, load_b_sw(sWd + l * 4096 + ((4 + nt) << 9), lane), c);
#pragma unroll
                for (int r = 0; r < 8; ++r) {
                    int row = r + (hl << 3);
                    int dcol = nt * 16 + m;
                    out[(long)(n0 + row) * 576 + lbase[l] + dcol * d + comp] = c[r] * scale;
                }
            }
        }
    }
}

// ---------------------------------------------------------------------------
extern "C" void kernel_launch(void* const* d_in, const int* in_sizes, int n_in,
                              void* d_out, int out_size, void* d_ws, size_t ws_size,
                              hipStream_t stream) {
    const float* edge_vectors = (const float*)d_in[0];
    const float* node_feats   = (const float*)d_in[1];
    const float* radial       = (const float*)d_in[2];
    const int*   senders      = (const int*)d_in[3];
    const int*   receivers    = (const int*)d_in[4];
    const float* W_up         = (const float*)d_in[5];
    const float* W_mlp1       = (const float*)d_in[6];
    const float* W_mlp2       = (const float*)d_in[7];
    const float* W_mlp3       = (const float*)d_in[8];
    const float* W_mlp4       = (const float*)d_in[9];
    const float* W_down       = (const float*)d_in[10];
    float* out = (float*)d_out;

    float* hG  = (float*)d_ws;                                     // N*64 f32
    float* agg = (float*)((char*)d_ws + (size_t)N_NODES * 64 * 4); // 9 planes of N*64 f32

    zero_kernel<<<2048, 256, 0, stream>>>(agg, 9 * PLANE);

    up_kernel<<<(N_NODES / 16 + 7) / 8, 256, 0, stream>>>(node_feats, W_up, hG);

    edge_kernel<<<N_EDGES / 64, 128, 0, stream>>>(edge_vectors, radial, senders,
                                                  receivers, W_mlp1, W_mlp2, W_mlp3,
                                                  W_mlp4, hG, agg);

    down_kernel<<<(N_NODES / 16 + 7) / 8, 256, 0, stream>>>(agg, W_down, out);
}
